// Attention_48155173322896
// MI455X (gfx1250) — matrix operand order repaired and checked by reading, not compiled, not run on hardware
//
#include <hip/hip_runtime.h>

typedef __attribute__((ext_vector_type(16))) _Float16 v16h;
typedef __attribute__((ext_vector_type(8)))  float    v8f;

#define BB 2
#define SS 2048
#define HH 32
#define KVH 8
#define DD 128
#define GQA (HH / KVH)
#define TT (BB * SS)
#define NUM_SLOTS 8192
#define SCALE 0.08838834764831845f  /* 1/sqrt(128) */

#define BLOCK_M 128
#define BLOCK_N 32
#define KPAD 8   /* K row = 136 halves = 272B (16B-aligned rows) */
#define VPAD 8   /* Vt row = 40 halves = 80B (16B-aligned rows) */
#define PPAD 8   /* P row  = 40 halves = 80B (16B-aligned rows) */

// Async global -> LDS copy of 16 bytes per lane (gfx1250, ASYNCcnt-tracked).
__device__ __forceinline__ void async_copy_b128(unsigned lds_off, const float* gptr)
{
    asm volatile("global_load_async_to_lds_b128 %0, %1, off"
                 :: "v"(lds_off), "v"((unsigned long long)(uintptr_t)gptr)
                 : "memory");
}
__device__ __forceinline__ void wait_async0()
{
    asm volatile("s_wait_asynccnt 0x0" ::: "memory");
}

// ---------------------------------------------------------------------------
// Flash attention, causal, GQA. One WG = (b, h, 128-query tile); 8 waves,
// each wave owns a 16-row query tile. f16 WMMA, f32 accumulate.
// K/V tiles are double-buffered: async DMA (f32 stage) overlapped with WMMA.
// ---------------------------------------------------------------------------
__global__ __launch_bounds__(256)
void fa_kernel(const float* __restrict__ q, const float* __restrict__ k,
               const float* __restrict__ v, float* __restrict__ out)
{
    __shared__ float    Kstage[BLOCK_N][DD];             // raw f32 K tile (async dest)
    __shared__ float    Vstage[BLOCK_N][DD];             // raw f32 V tile (async dest)
    __shared__ _Float16 Klds[BLOCK_N][DD + KPAD];        // K tile, f16 [key][d]
    __shared__ _Float16 Vtlds[DD][BLOCK_N + VPAD];       // V tile, f16 transposed [d][key]
    __shared__ _Float16 Plds[8][16][BLOCK_N + PPAD];     // per-wave P staging [m][k]

    const int tid    = threadIdx.x;
    const int wave   = tid >> 5;
    const int lane   = tid & 31;
    const int lane16 = lane & 15;
    const int grp    = lane >> 4;

    const int qt  = blockIdx.x;
    const int h   = blockIdx.y;
    const int b   = blockIdx.z;
    const int kvh = h / GQA;

    const int qbase_blk  = qt * BLOCK_M;
    const int qbase_wave = qbase_blk + wave * 16;

    const unsigned kstage0 = (unsigned)(uintptr_t)&Kstage[0][0];
    const unsigned vstage0 = (unsigned)(uintptr_t)&Vstage[0][0];

    // ---- Q fragments in WMMA A-layout, pre-scaled by 1/sqrt(D) ----
    // A 16x32 f16: lanes 0-15 row M=lane, K={0..7,16..23}; lanes 16-31 same M, K+8.
    v16h aQ[4];
    {
        const int qm = qbase_wave + lane16;
        const float* qrow = q + (size_t)(b * SS + qm) * (HH * DD) + h * DD;
        #pragma unroll
        for (int ks = 0; ks < 4; ++ks) {
            #pragma unroll
            for (int e = 0; e < 16; ++e) {
                int vreg = e >> 1, lo = e & 1;
                int dk = (vreg < 4 ? 2 * vreg : 16 + 2 * (vreg - 4)) + lo + grp * 8;
                aQ[ks][e] = (_Float16)(qrow[ks * 32 + dk] * SCALE);
            }
        }
    }

    v8f zacc = {};
    v8f o_acc[8];
    float m_run[8], l_run[8];
    #pragma unroll
    for (int i = 0; i < 8; ++i) { o_acc[i] = zacc; m_run[i] = -1e30f; l_run[i] = 0.0f; }

    const int nkb_blk  = (qbase_blk + BLOCK_M) / BLOCK_N;   // key blocks for the WG
    const int kb_max_w = (qbase_wave + 15) / BLOCK_N;       // inclusive limit, this wave

    // Issue async DMA of one 32x128 f32 K/V tile pair: 8 x b128 chunks per thread.
    auto issue_tile = [&](int kbn) {
        const int kabs0 = kbn * BLOCK_N;
        #pragma unroll
        for (int c = 0; c < 4; ++c) {
            int idx = tid + c * 256;                 // 16-byte chunk id (0..1023)
            int f   = idx << 2;                      // flat float index
            int key = f >> 7;                        // / DD
            int d   = f & (DD - 1);
            size_t g = (size_t)(b * SS + kabs0 + key) * (KVH * DD) + kvh * DD + d;
            async_copy_b128(kstage0 + idx * 16, k + g);
            async_copy_b128(vstage0 + idx * 16, v + g);
        }
    };

    issue_tile(0);   // prologue

    for (int kb = 0; kb < nkb_blk; ++kb) {
        wait_async0();          // my async writes to stage are complete
        __syncthreads();        // everyone's are; prev-iter f16-tile reads done too

        // ---- convert staged f32 tile -> f16 tiles (V transposed) ----
        for (int idx = tid; idx < BLOCK_N * DD; idx += 256) {
            int key = idx >> 7;
            int d   = idx & (DD - 1);
            Klds[key][d]  = (_Float16)Kstage[key][d];
            Vtlds[d][key] = (_Float16)Vstage[key][d];
        }
        __syncthreads();        // f16 tiles published; stage buffers now reusable

        if (kb + 1 < nkb_blk) issue_tile(kb + 1);   // DMA overlapped with WMMAs below

        if (kb <= kb_max_w) {   // wave-uniform predicate; EXEC all-ones inside
            // ---- GEMM1: S = (Q*scale) @ K^T, two 16x16 N-tiles, K=128 in 4 steps ----
            v8f s_acc[2];
            s_acc[0] = zacc; s_acc[1] = zacc;
            #pragma unroll
            for (int nt = 0; nt < 2; ++nt) {
                #pragma unroll
                for (int ks = 0; ks < 4; ++ks) {
                    v16h bK;
                    #pragma unroll
                    for (int e = 0; e < 16; ++e) {
                        int vreg = e >> 1, lo = e & 1;
                        int dk = (vreg < 4 ? 2 * vreg : 16 + 2 * (vreg - 4)) + lo + grp * 8;   // K index in 32-chunk
                        bK[e] = Klds[nt * 16 + lane16][ks * 32 + dk];
                    }
                    s_acc[nt] = __builtin_amdgcn_wmma_f32_16x16x32_f16(
                        false, aQ[ks], false, bK, (short)0, s_acc[nt], false, false);
                }
            }

            // ---- causal mask + online softmax (rows live in 16-lane halves) ----
            float tmax[8];
            #pragma unroll
            for (int r = 0; r < 8; ++r) {
                int qrow_r = qbase_wave + r + 8 * grp;
                #pragma unroll
                for (int nt = 0; nt < 2; ++nt) {
                    int kcol = kb * BLOCK_N + nt * 16 + lane16;
                    if (kcol > qrow_r) s_acc[nt][r] = -1e30f;
                }
                float mv = fmaxf(s_acc[0][r], s_acc[1][r]);
                #pragma unroll
                for (int msk = 1; msk <= 8; msk <<= 1)
                    mv = fmaxf(mv, __shfl_xor(mv, msk, 32));
                tmax[r] = mv;
            }

            #pragma unroll
            for (int r = 0; r < 8; ++r) {
                float mnew  = fmaxf(m_run[r], tmax[r]);
                float alpha = __expf(m_run[r] - mnew);
                float p0 = __expf(s_acc[0][r] - mnew);
                float p1 = __expf(s_acc[1][r] - mnew);
                s_acc[0][r] = p0; s_acc[1][r] = p1;
                float sv = p0 + p1;
                #pragma unroll
                for (int msk = 1; msk <= 8; msk <<= 1)
                    sv += __shfl_xor(sv, msk, 32);
                l_run[r] = l_run[r] * alpha + sv;
                m_run[r] = mnew;
                #pragma unroll
                for (int dt = 0; dt < 8; ++dt) o_acc[dt][r] *= alpha;
            }

            // ---- stage P (C-layout -> row-major f16 in LDS, wave-private) ----
            #pragma unroll
            for (int nt = 0; nt < 2; ++nt)
                #pragma unroll
                for (int r = 0; r < 8; ++r)
                    Plds[wave][r + 8 * grp][nt * 16 + lane16] = (_Float16)s_acc[nt][r];

            // ---- GEMM2: O += P(16x32) @ V(32x128), 8 d-tiles ----
            v16h aP;
            #pragma unroll
            for (int e = 0; e < 16; ++e) {
                int vreg = e >> 1, lo = e & 1;
                int kk = (vreg < 4 ? 2 * vreg : 16 + 2 * (vreg - 4)) + lo + grp * 8;
                aP[e] = Plds[wave][lane16][kk];
            }
            #pragma unroll
            for (int dt = 0; dt < 8; ++dt) {
                v16h bV;
                #pragma unroll
                for (int e = 0; e < 16; ++e) {
                    int vreg = e >> 1, lo = e & 1;
                    int kk = (vreg < 4 ? 2 * vreg : 16 + 2 * (vreg - 4)) + lo + grp * 8;       // key index 0..31
                    bV[e] = Vtlds[dt * 16 + lane16][kk];
                }
                o_acc[dt] = __builtin_amdgcn_wmma_f32_16x16x32_f16(
                    false, aP, false, bV, (short)0, o_acc[dt], false, false);
            }
        }
    }

    // ---- normalize and store ----
    #pragma unroll
    for (int r = 0; r < 8; ++r) {
        int qrow_r = qbase_wave + r + 8 * grp;
        float inv = 1.0f / l_run[r];
        float* orow = out + (size_t)(b * SS + qrow_r) * (HH * DD) + h * DD;
        #pragma unroll
        for (int dt = 0; dt < 8; ++dt)
            orow[dt * 16 + lane16] = o_acc[dt][r] * inv;
    }
}

// ---------------------------------------------------------------------------
// Cache copy (d_out is poisoned; every byte of kc/vc outputs must be written)
// ---------------------------------------------------------------------------
__global__ void copy_cache_kernel(const float4* __restrict__ kc_in,
                                  const float4* __restrict__ vc_in,
                                  float4* __restrict__ kc_out,
                                  float4* __restrict__ vc_out, int n4)
{
    int i = blockIdx.x * blockDim.x + threadIdx.x;
    if (i < n4) { kc_out[i] = kc_in[i]; vc_out[i] = vc_in[i]; }
}

// Scatter current k/v rows into the caches by slot_mapping (int64 indices).
__global__ void scatter_kv_kernel(const float* __restrict__ k,
                                  const float* __restrict__ v,
                                  const long long* __restrict__ slot_mapping,
                                  float* __restrict__ kc_out,
                                  float* __restrict__ vc_out)
{
    int t = blockIdx.x;
    long long slot = slot_mapping[t];
    int c = threadIdx.x;                       // 0..255 float4 chunks of 1024 floats
    const float4* ksrc = (const float4*)(k + (size_t)t * (KVH * DD));
    const float4* vsrc = (const float4*)(v + (size_t)t * (KVH * DD));
    float4* kdst = (float4*)(kc_out + (size_t)slot * (KVH * DD));
    float4* vdst = (float4*)(vc_out + (size_t)slot * (KVH * DD));
    kdst[c] = ksrc[c];
    vdst[c] = vsrc[c];
}

extern "C" void kernel_launch(void* const* d_in, const int* in_sizes, int n_in,
                              void* d_out, int out_size, void* d_ws, size_t ws_size,
                              hipStream_t stream)
{
    const float* q  = (const float*)d_in[0];
    const float* k  = (const float*)d_in[1];
    const float* v  = (const float*)d_in[2];
    const float* kc = (const float*)d_in[3];
    const float* vc = (const float*)d_in[4];
    const long long* slot = (const long long*)d_in[5];

    float* out_attn = (float*)d_out;
    float* out_kc = out_attn + (size_t)TT * HH * DD;
    float* out_vc = out_kc + (size_t)NUM_SLOTS * KVH * DD;

    int n4 = NUM_SLOTS * KVH * DD / 4;
    copy_cache_kernel<<<(n4 + 255) / 256, 256, 0, stream>>>(
        (const float4*)kc, (const float4*)vc, (float4*)out_kc, (float4*)out_vc, n4);
    scatter_kv_kernel<<<TT, 256, 0, stream>>>(k, v, slot, out_kc, out_vc);

    dim3 grid(SS / BLOCK_M, HH, BB);
    fa_kernel<<<grid, 256, 0, stream>>>(q, k, v, out_attn);
}
